// AttentionBlock_29549374996876
// MI455X (gfx1250) — compile-verified, matrix operand order checked by reading, not run on hardware
//
#include <hip/hip_runtime.h>
#include <hip/hip_bf16.h>

// ---------------- problem constants ----------------
#define S_LEN   1536
#define HID     2880
#define N_HEADS 64
#define N_KV    8
#define D_HEAD  64
#define QKV_DIM 5120           // 64*64 + 2*8*64
#define O_DIM   4096           // 64*64
#define EPSV    1e-5f
#define SM_SCALE 0.125f        // 1/sqrt(64)
#define ROPE_BASE 150000.0f
#define INIT_CTX 4096.0f
#define SCALING  32.0f
#define NTK_ALPHA 1.0f
#define NTK_BETA  32.0f

typedef __attribute__((ext_vector_type(16))) __bf16 v16bf;
typedef __attribute__((ext_vector_type(8)))  float  v8f;

// Scheduling fence: memory ops may not be reordered across it.  Used to pin
// fragment-load batches above the WMMA groups that consume the *other*
// pipeline stage, so the scheduler cannot sink loads to just before their
// consumers (which serializes behind s_wait_loadcnt 0).
#define SCHED_FENCE() asm volatile("" ::: "memory")

// ---------------- bf16 helpers ----------------
__device__ __forceinline__ unsigned short f2bf(float f) {
    union { float f; unsigned u; } c; c.f = f;
    unsigned u = c.u;
    u += 0x7FFFu + ((u >> 16) & 1u);   // round-to-nearest-even
    return (unsigned short)(u >> 16);
}

union Frag {            // 16 bf16 = 32 bytes = 8 VGPRs (wave32)
    v16bf v;
    unsigned short u[16];
    uint4 q[2];
};

// Load a 16-element bf16 fragment from row-major [row][K] memory.
// Per CDNA5 ISA 7.12.2, element i maps to K = kbase + i + 8*(i>=8) + 8*(lane>=16),
// i.e. two contiguous 16-byte chunks at kbase+off and kbase+off+16 (off = hi?8:0).
__device__ __forceinline__ v16bf load_frag(const unsigned short* __restrict__ rowptr,
                                           int kbase, int hi) {
    Frag f;
    int off = hi ? 8 : 0;
    f.q[0] = *reinterpret_cast<const uint4*>(rowptr + kbase + off);
    f.q[1] = *reinterpret_cast<const uint4*>(rowptr + kbase + off + 16);
    return f.v;
}

__device__ __forceinline__ v8f wmma_bf16(v16bf a, v16bf b, v8f c) {
    return __builtin_amdgcn_wmma_f32_16x16x32_bf16(false, a, false, b,
                                                   (short)0, c, false, false);
}

// ---------------- kernel 1: f32 -> bf16 conversion ----------------
__global__ void cvt_f32_bf16_kernel(const float* __restrict__ src,
                                    unsigned short* __restrict__ dst, long n) {
    long i = (long)blockIdx.x * blockDim.x + threadIdx.x;
    long stride = (long)gridDim.x * blockDim.x;
    for (; i < n; i += stride) dst[i] = f2bf(src[i]);
}

// ---------------- kernel 2: RMSNorm -> bf16 ----------------
__global__ void rmsnorm_kernel(const float* __restrict__ x,
                               const float* __restrict__ gamma,
                               unsigned short* __restrict__ t) {
    __shared__ float red[256];
    int s = blockIdx.x, tid = threadIdx.x;
    const float* row = x + (size_t)s * HID;
    float ss = 0.f;
    for (int i = tid; i < HID; i += 256) { float v = row[i]; ss += v * v; }
    red[tid] = ss; __syncthreads();
    for (int o = 128; o > 0; o >>= 1) {
        if (tid < o) red[tid] += red[tid + o];
        __syncthreads();
    }
    float inv = rsqrtf(red[0] / (float)HID + EPSV);
    unsigned short* trow = t + (size_t)s * HID;
    for (int i = tid; i < HID; i += 256) trow[i] = f2bf(row[i] * inv * gamma[i]);
}

// ---------------- wave-level 16x64 bf16 GEMM core ----------------
// Ping-pong software pipeline, unrolled by 2 (requires K % 64 == 0).
// Two fragment sets alternate; SCHED_FENCE pins each load batch above the
// other stage's WMMA group so fragments stay in flight across the WMMAs
// instead of being sunk next to their consumers.
__device__ __forceinline__ void wave_gemm_16x64(
    const unsigned short* __restrict__ A, int lda, int mbase,
    const unsigned short* __restrict__ B, int ldb, int nbase,
    int K, v8f acc[4]) {
    int lane = threadIdx.x & 31;
    int hi = lane >> 4, lm = lane & 15;
    const unsigned short* pa = A + (size_t)(mbase + lm) * lda;
    const unsigned short* p0 = B + (size_t)(nbase + 0 * 16 + lm) * ldb;
    const unsigned short* p1 = B + (size_t)(nbase + 1 * 16 + lm) * ldb;
    const unsigned short* p2 = B + (size_t)(nbase + 2 * 16 + lm) * ldb;
    const unsigned short* p3 = B + (size_t)(nbase + 3 * 16 + lm) * ldb;

    // set 0: fragments for K-chunk [k, k+32)
    v16bf a0  = load_frag(pa, 0, hi);
    v16bf b00 = load_frag(p0, 0, hi);
    v16bf b01 = load_frag(p1, 0, hi);
    v16bf b02 = load_frag(p2, 0, hi);
    v16bf b03 = load_frag(p3, 0, hi);

    for (int k = 0; k < K; k += 64) {
        // set 1 loads for [k+32, k+64) — always in range since K % 64 == 0
        v16bf a1  = load_frag(pa, 32, hi);
        v16bf b10 = load_frag(p0, 32, hi);
        v16bf b11 = load_frag(p1, 32, hi);
        v16bf b12 = load_frag(p2, 32, hi);
        v16bf b13 = load_frag(p3, 32, hi);
        SCHED_FENCE();                      // keep set-1 loads above set-0 WMMAs
        acc[0] = wmma_bf16(a0, b00, acc[0]);
        acc[1] = wmma_bf16(a0, b01, acc[1]);
        acc[2] = wmma_bf16(a0, b02, acc[2]);
        acc[3] = wmma_bf16(a0, b03, acc[3]);
        pa += 64; p0 += 64; p1 += 64; p2 += 64; p3 += 64;
        if (k + 64 < K) {           // set 0 reload for [k+64, k+96)
            a0  = load_frag(pa, 0, hi);
            b00 = load_frag(p0, 0, hi);
            b01 = load_frag(p1, 0, hi);
            b02 = load_frag(p2, 0, hi);
            b03 = load_frag(p3, 0, hi);
        }
        SCHED_FENCE();                      // keep set-0 reloads above set-1 WMMAs
        acc[0] = wmma_bf16(a1, b10, acc[0]);
        acc[1] = wmma_bf16(a1, b11, acc[1]);
        acc[2] = wmma_bf16(a1, b12, acc[2]);
        acc[3] = wmma_bf16(a1, b13, acc[3]);
    }
}

// ---------------- kernel 3: QKV GEMM (bf16 WMMA, f32 out + bias) ----------------
// grid (QKV_DIM/256, S/32), block 256 (8 waves: 2 M-tiles x 4 N-groups)
__global__ void __launch_bounds__(256, 2)
qkv_gemm_kernel(const unsigned short* __restrict__ t,
                const unsigned short* __restrict__ wq,
                const float* __restrict__ bias,
                float* __restrict__ qkv) {
    int w = threadIdx.x >> 5;
    int mbase = blockIdx.y * 32 + (w & 1) * 16;
    int nbase = blockIdx.x * 256 + (w >> 1) * 64;
    v8f acc[4] = {};
    wave_gemm_16x64(t, HID, mbase, wq, HID, nbase, HID, acc);
    int lane = threadIdx.x & 31, hi = lane >> 4, lm = lane & 15;
    for (int j = 0; j < 4; ++j) {
        int col = nbase + j * 16 + lm;
        float b = bias[col];
        for (int r = 0; r < 8; ++r) {
            int row = mbase + r + 8 * hi;
            qkv[(size_t)row * QKV_DIM + col] = acc[j][r] + b;
        }
    }
}

// ---------------- kernel 4: YaRN RoPE + scatter q/k/v to bf16 ----------------
// grid S, block 256.  V is written TRANSPOSED (vt[kv][d][s]) so attention's PV
// B-fragments become two contiguous b128 loads per lane.
__global__ void rope_kernel(const float* __restrict__ qkv,
                            unsigned short* __restrict__ q,   // [NH][S][D]
                            unsigned short* __restrict__ kk,  // [NKV][S][D]
                            unsigned short* __restrict__ vt) {// [NKV][D][S]
    int s = blockIdx.x, tid = threadIdx.x;
    const float* row = qkv + (size_t)s * QKV_DIM;
    const float lr = logf(ROPE_BASE);
    const float low  = 32.f * logf(INIT_CTX / (NTK_BETA  * 6.2831853f)) / lr;
    const float high = 32.f * logf(INIT_CTX / (NTK_ALPHA * 6.2831853f)) / lr;
    const float conc = 0.1f * logf(SCALING) + 1.0f;

    // Q: 64 heads x 32 rotation pairs
    for (int tloop = 0; tloop < 8; ++tloop) {
        int pi = tid + tloop * 256;          // 0..2047
        int h = pi >> 5, j = pi & 31;
        float freq = powf(ROPE_BASE, (float)j / 32.f);
        float interp = 1.f / (SCALING * freq), extrap = 1.f / freq;
        float ramp = ((float)j - low) / (high - low);
        float mask = 1.f - fminf(fmaxf(ramp, 0.f), 1.f);
        float invf = interp * (1.f - mask) + extrap * mask;
        float ang = (float)s * invf;
        float c = cosf(ang) * conc, sn = sinf(ang) * conc;
        float x1 = row[h * D_HEAD + j], x2 = row[h * D_HEAD + j + 32];
        unsigned short* qrow = q + ((size_t)h * S_LEN + s) * D_HEAD;
        qrow[j]      = f2bf((x1 * c - x2 * sn) * SM_SCALE);
        qrow[j + 32] = f2bf((x2 * c + x1 * sn) * SM_SCALE);
    }
    // K: 8 heads x 32 pairs
    {
        int kvh = tid >> 5, j = tid & 31;
        float freq = powf(ROPE_BASE, (float)j / 32.f);
        float interp = 1.f / (SCALING * freq), extrap = 1.f / freq;
        float ramp = ((float)j - low) / (high - low);
        float mask = 1.f - fminf(fmaxf(ramp, 0.f), 1.f);
        float invf = interp * (1.f - mask) + extrap * mask;
        float ang = (float)s * invf;
        float c = cosf(ang) * conc, sn = sinf(ang) * conc;
        const float* kr = row + O_DIM + kvh * D_HEAD;
        float x1 = kr[j], x2 = kr[j + 32];
        unsigned short* krow = kk + ((size_t)kvh * S_LEN + s) * D_HEAD;
        krow[j]      = f2bf(x1 * c - x2 * sn);
        krow[j + 32] = f2bf(x2 * c + x1 * sn);
    }
    // V: transpose-scatter 512 elements -> vt[kv][d][s]
    for (int tloop = 0; tloop < 2; ++tloop) {
        int e = tid + tloop * 256;           // 0..511
        int kvh = e >> 6, d = e & 63;
        vt[((size_t)kvh * D_HEAD + d) * S_LEN + s] =
            f2bf(row[O_DIM + N_KV * D_HEAD + e]);
    }
}

// ---------------- kernel 5: flash attention, one wave per (qtile, head) ----------------
// grid (S/16, NH), block 32
__global__ void attn_kernel(const unsigned short* __restrict__ q,
                            const unsigned short* __restrict__ kk,
                            const unsigned short* __restrict__ vt,
                            const float* __restrict__ sinks,
                            unsigned short* __restrict__ ao) { // [S][O_DIM] bf16
    __shared__ unsigned short pBf[16 * 32];   // P tile, bf16 row-major [m][k]
    int qt = blockIdx.x, h = blockIdx.y;
    int kvh = h & (N_KV - 1);                 // reference: kv index = head % 8
    int qbase = qt * 16;
    int lane = threadIdx.x & 31, hi = lane >> 4, lm = lane & 15;

    const unsigned short* qrow = q + ((size_t)h * S_LEN + qbase + lm) * D_HEAD;
    v16bf aq0 = load_frag(qrow, 0, hi);
    v16bf aq1 = load_frag(qrow, 32, hi);

    // V row pointers (transposed layout: [kv][d][s]), d = c4*16 + lm per lane
    const unsigned short* vrow[4];
    for (int c4 = 0; c4 < 4; ++c4)
        vrow[c4] = vt + ((size_t)kvh * D_HEAD + c4 * 16 + lm) * S_LEN;

    v8f acc[4] = {};
    float m_run[8], l_run[8];
    for (int r = 0; r < 8; ++r) { m_run[r] = -1e30f; l_run[r] = 0.f; }

    int nblk = ((qbase + 15) >> 5) + 1;       // 32-wide kv blocks
    for (int jb = 0; jb < nblk; ++jb) {
        int kb = jb * 32;
        // ---- issue ALL global loads for this block up front ----
        const unsigned short* krow0 = kk + ((size_t)kvh * S_LEN + kb + lm) * D_HEAD;
        const unsigned short* krow1 = krow0 + (size_t)16 * D_HEAD;
        v16bf kf00 = load_frag(krow0, 0, hi);   // sub 0, d 0..31
        v16bf kf01 = load_frag(krow0, 32, hi);  // sub 0, d 32..63
        v16bf kf10 = load_frag(krow1, 0, hi);   // sub 1
        v16bf kf11 = load_frag(krow1, 32, hi);
        v16bf vf[4];
        for (int c4 = 0; c4 < 4; ++c4)          // arrive during softmax math
            vf[c4] = load_frag(vrow[c4], kb, hi);
        SCHED_FENCE();

        // ---- scores: two 16x16 subtiles over kv cols [kb, kb+32) ----
        v8f sc[2];
        {
            v8f c = {};
            c = wmma_bf16(aq0, kf00, c);
            c = wmma_bf16(aq1, kf01, c);
            sc[0] = c;
            v8f d = {};
            d = wmma_bf16(aq0, kf10, d);
            d = wmma_bf16(aq1, kf11, d);
            sc[1] = d;
        }
        for (int sub = 0; sub < 2; ++sub) {
            int col = kb + sub * 16 + lm;
            for (int r = 0; r < 8; ++r) {
                int row = qbase + r + 8 * hi;
                if (col > row) sc[sub][r] = -1e30f;   // causal mask
            }
        }
        // ---- running softmax stats (row lives in a 16-lane half) ----
        float tmax[8];
        for (int r = 0; r < 8; ++r) tmax[r] = fmaxf(sc[0][r], sc[1][r]);
        for (int msk = 1; msk < 16; msk <<= 1)
            for (int r = 0; r < 8; ++r)
                tmax[r] = fmaxf(tmax[r], __shfl_xor(tmax[r], msk, 32));
        float corr[8], mnew[8];
        for (int r = 0; r < 8; ++r) {
            mnew[r] = fmaxf(m_run[r], tmax[r]);
            corr[r] = __expf(m_run[r] - mnew[r]);
        }
        float lsum[8];
        for (int r = 0; r < 8; ++r) {
            sc[0][r] = __expf(sc[0][r] - mnew[r]);
            sc[1][r] = __expf(sc[1][r] - mnew[r]);
            lsum[r] = sc[0][r] + sc[1][r];
        }
        for (int msk = 1; msk < 16; msk <<= 1)
            for (int r = 0; r < 8; ++r)
                lsum[r] += __shfl_xor(lsum[r], msk, 32);
        for (int r = 0; r < 8; ++r) {
            l_run[r] = l_run[r] * corr[r] + lsum[r];
            m_run[r] = mnew[r];
        }
        for (int c4 = 0; c4 < 4; ++c4)
            for (int r = 0; r < 8; ++r) acc[c4][r] *= corr[r];
        // ---- re-layout P: C-layout -> LDS(bf16) -> A-layout via 2x ds_load_b128 ----
        for (int sub = 0; sub < 2; ++sub)
            for (int r = 0; r < 8; ++r)
                pBf[(r + 8 * hi) * 32 + sub * 16 + lm] = f2bf(sc[sub][r]);
        asm volatile("s_wait_dscnt 0" ::: "memory");
        Frag pf;
        {
            int off = hi ? 8 : 0;
            pf.q[0] = *reinterpret_cast<const uint4*>(&pBf[lm * 32 + off]);
            pf.q[1] = *reinterpret_cast<const uint4*>(&pBf[lm * 32 + off + 16]);
        }
        // ---- PV: 4 d-chunks of 16 (V fragments already resident) ----
        for (int c4 = 0; c4 < 4; ++c4)
            acc[c4] = wmma_bf16(pf.v, vf[c4], acc[c4]);
        asm volatile("s_wait_dscnt 0" ::: "memory"); // LDS reuse next iter
    }
    // ---- sink correction + normalize + store ----
    float snk = sinks[h];
    float scl[8];
    for (int r = 0; r < 8; ++r) {
        float mf = fmaxf(m_run[r], snk);
        float e  = __expf(m_run[r] - mf);
        float lf = l_run[r] * e + __expf(snk - mf);
        scl[r] = e / lf;
    }
    for (int c4 = 0; c4 < 4; ++c4)
        for (int r = 0; r < 8; ++r) {
            int row = qbase + r + 8 * hi;
            ao[(size_t)row * O_DIM + h * D_HEAD + c4 * 16 + lm] =
                f2bf(acc[c4][r] * scl[r]);
        }
}

// ---------------- kernel 6: out GEMM + bias + residual ----------------
// grid (ceil(HID/256)=12, S/32), block 256
__global__ void __launch_bounds__(256, 2)
out_gemm_kernel(const unsigned short* __restrict__ ao,
                const unsigned short* __restrict__ wo,
                const float* __restrict__ bias,
                const float* __restrict__ x,
                float* __restrict__ out) {
    int w = threadIdx.x >> 5;
    int mbase = blockIdx.y * 32 + (w & 1) * 16;
    int nbase = blockIdx.x * 256 + (w >> 1) * 64;
    if (nbase >= HID) return;
    v8f acc[4] = {};
    wave_gemm_16x64(ao, O_DIM, mbase, wo, O_DIM, nbase, O_DIM, acc);
    int lane = threadIdx.x & 31, hi = lane >> 4, lm = lane & 15;
    for (int j = 0; j < 4; ++j) {
        int col = nbase + j * 16 + lm;
        float b = bias[col];
        for (int r = 0; r < 8; ++r) {
            int row = mbase + r + 8 * hi;
            out[(size_t)row * HID + col] =
                acc[j][r] + b + x[(size_t)row * HID + col];
        }
    }
}

// ---------------- launch ----------------
static inline size_t alignup(size_t v) { return (v + 255) & ~(size_t)255; }

extern "C" void kernel_launch(void* const* d_in, const int* in_sizes, int n_in,
                              void* d_out, int out_size, void* d_ws, size_t ws_size,
                              hipStream_t stream) {
    const float* x          = (const float*)d_in[0];
    const float* sinks      = (const float*)d_in[1];
    const float* norm_scale = (const float*)d_in[2];
    const float* qkv_w      = (const float*)d_in[3];
    const float* qkv_b      = (const float*)d_in[4];
    const float* out_w      = (const float*)d_in[5];
    const float* out_b      = (const float*)d_in[6];
    float* out              = (float*)d_out;

    char* ws = (char*)d_ws;
    size_t off = 0;
    unsigned short* t_bf  = (unsigned short*)(ws + off); off = alignup(off + (size_t)S_LEN * HID * 2);
    unsigned short* wq_bf = (unsigned short*)(ws + off); off = alignup(off + (size_t)QKV_DIM * HID * 2);
    unsigned short* wo_bf = (unsigned short*)(ws + off); off = alignup(off + (size_t)HID * O_DIM * 2);
    float*          qkvf  = (float*)(ws + off);          off = alignup(off + (size_t)S_LEN * QKV_DIM * 4);
    unsigned short* q_bf  = (unsigned short*)(ws + off); off = alignup(off + (size_t)N_HEADS * S_LEN * D_HEAD * 2);
    unsigned short* k_bf  = (unsigned short*)(ws + off); off = alignup(off + (size_t)N_KV * S_LEN * D_HEAD * 2);
    unsigned short* vt_bf = (unsigned short*)(ws + off); off = alignup(off + (size_t)N_KV * S_LEN * D_HEAD * 2);
    unsigned short* ao_bf = (unsigned short*)(ws + off); off = alignup(off + (size_t)S_LEN * O_DIM * 2);
    (void)ws_size; (void)n_in; (void)in_sizes; (void)out_size;

    cvt_f32_bf16_kernel<<<1024, 256, 0, stream>>>(qkv_w, wq_bf, (long)QKV_DIM * HID);
    cvt_f32_bf16_kernel<<<1024, 256, 0, stream>>>(out_w, wo_bf, (long)HID * O_DIM);
    rmsnorm_kernel<<<S_LEN, 256, 0, stream>>>(x, norm_scale, t_bf);
    qkv_gemm_kernel<<<dim3(QKV_DIM / 256, S_LEN / 32), 256, 0, stream>>>(t_bf, wq_bf, qkv_b, qkvf);
    rope_kernel<<<S_LEN, 256, 0, stream>>>(qkvf, q_bf, k_bf, vt_bf);
    attn_kernel<<<dim3(S_LEN / 16, N_HEADS), 32, 0, stream>>>(q_bf, k_bf, vt_bf, sinks, ao_bf);
    out_gemm_kernel<<<dim3((HID + 255) / 256, S_LEN / 32), 256, 0, stream>>>(ao_bf, wo_bf, out_b, x, out);
}
